// LayerGCN_36850819399940
// MI455X (gfx1250) — matrix operand order repaired and checked by reading, not compiled, not run on hardware
//
#include <hip/hip_runtime.h>
#include <hip/hip_bf16.h>
#include <math.h>

#define N_USER 100000
#define N_ITEM 50000
#define N_TOT  (N_USER + N_ITEM)
#define EMB    64
#define FEAT   384
#define NNZ    2400000
#define N_PROMPT 8
#define N_LAYERS 4

typedef __attribute__((ext_vector_type(16))) _Float16 v16h;
typedef __attribute__((ext_vector_type(8)))  _Float16 v8h;
typedef __attribute__((ext_vector_type(8)))  float    v8f;

// ---------------------------------------------------------------------------
// Stage 1: user rows of ego:  ego[r][c] = user_fea[r][c] + sum_p prompt[p][c]
// ---------------------------------------------------------------------------
__global__ void lgcn_user_embed(const float* __restrict__ user_fea,
                                const float* __restrict__ prompt,
                                float* __restrict__ ego) {
    int idx = blockIdx.x * blockDim.x + threadIdx.x;
    if (idx >= N_USER * EMB) return;
    int c = idx & (EMB - 1);
    float p = 0.f;
#pragma unroll
    for (int j = 0; j < N_PROMPT; ++j) p += prompt[j * EMB + c];   // L0-cached
    ego[idx] = user_fea[idx] + p;
}

// ---------------------------------------------------------------------------
// Stage 2: item rows of ego via WMMA:  ego_item = tanh(item_fea @ mlp_w + b)
// Block = 256 threads = 8 waves; each wave computes a 16x64 output slab.
//
// mlp_w is staged into LDS as f16 in *WMMA B-fragment order*:
//   wlds[((kblk*2 + khalf)*64 + n)*16 + i] = f16(mlp_w[(kblk*32+khalf*16+i)*64 + n])
// so each lane's 16-half fragment is 32 contiguous bytes -> two ds_load_b128,
// no scalar ds_load_u16 / v_mov_b16 repacking.
// ---------------------------------------------------------------------------
__global__ void lgcn_item_mlp_wmma(const float* __restrict__ item_fea,
                                   const float* __restrict__ mlp_w,
                                   const float* __restrict__ mlp_b,
                                   float* __restrict__ ego_item) {
    __shared__ __align__(32) _Float16 wlds[FEAT * EMB];            // 48 KB
    for (int idx = threadIdx.x; idx < FEAT * EMB; idx += blockDim.x) {
        int i    = idx & 15;
        int n    = (idx >> 4) & 63;
        int kh   = (idx >> 10) & 1;
        int kblk = idx >> 11;
        int k    = kblk * 32 + kh * 16 + i;
        wlds[idx] = (_Float16)mlp_w[k * EMB + n];
    }
    __syncthreads();

    const int wave  = threadIdx.x >> 5;
    const int lane  = threadIdx.x & 31;
    const int lrow  = lane & 15;          // M (A) / N (B,C,D) position
    const int khalf = lane >> 4;          // half-wave selector
    const int m0    = (blockIdx.x * 8 + wave) * 16;
    if (m0 >= N_ITEM) return;

    const float* __restrict__ arow = item_fea + (size_t)(m0 + lrow) * FEAT;

    v8f acc[4] = {};                      // 4 N-tiles of 16 -> 64 cols
    for (int kb = 0; kb < FEAT; kb += 32) {
        // A fragment, 16x32 f16: element i -> K = kb + khalf*16 + i (matches B).
        // Contiguous -> float4 loads.
        const float4 a0 = *(const float4*)(arow + kb + khalf * 16);
        const float4 a1 = *(const float4*)(arow + kb + khalf * 16 + 4);
        const float4 a2 = *(const float4*)(arow + kb + khalf * 16 + 8);
        const float4 a3 = *(const float4*)(arow + kb + khalf * 16 + 12);
        v16h afrag;
        afrag[0]  = (_Float16)a0.x; afrag[1]  = (_Float16)a0.y;
        afrag[2]  = (_Float16)a0.z; afrag[3]  = (_Float16)a0.w;
        afrag[4]  = (_Float16)a1.x; afrag[5]  = (_Float16)a1.y;
        afrag[6]  = (_Float16)a1.z; afrag[7]  = (_Float16)a1.w;
        afrag[8]  = (_Float16)a2.x; afrag[9]  = (_Float16)a2.y;
        afrag[10] = (_Float16)a2.z; afrag[11] = (_Float16)a2.w;
        afrag[12] = (_Float16)a3.x; afrag[13] = (_Float16)a3.y;
        afrag[14] = (_Float16)a3.z; afrag[15] = (_Float16)a3.w;

        const int kblk = kb >> 5;
#pragma unroll
        for (int nt = 0; nt < 4; ++nt) {
            const int n   = nt * 16 + lrow;
            const int off = ((kblk * 2 + khalf) * 64 + n) * 16;  // halves
            const v8h b0 = *(const v8h*)(wlds + off);            // ds_load_b128
            const v8h b1 = *(const v8h*)(wlds + off + 8);        // ds_load_b128
            const v16h bfrag = __builtin_shufflevector(
                b0, b1, 0, 1, 2, 3, 4, 5, 6, 7, 8, 9, 10, 11, 12, 13, 14, 15);
            acc[nt] = __builtin_amdgcn_wmma_f32_16x16x32_f16(
                /*neg_a=*/false, afrag, /*neg_b=*/false, bfrag,
                /*c_mod=*/(short)0, acc[nt],
                /*reuse_a=*/false, /*reuse_b=*/false);
        }
    }

    // C/D layout: VGPR r, lanes 0-15 -> M=r, lanes 16-31 -> M=8+r; N = lane&15
#pragma unroll
    for (int nt = 0; nt < 4; ++nt) {
        const int   col  = nt * 16 + lrow;
        const float bias = mlp_b[col];
#pragma unroll
        for (int r = 0; r < 8; ++r) {
            const int row = m0 + khalf * 8 + r;
            ego_item[(size_t)row * EMB + col] = tanhf(acc[nt][r] + bias);
        }
    }
}

// ---------------------------------------------------------------------------
// Stage 3: per-row norm of ego; seed x = ego, acc(d_out) = ego.
// One wave per row (wave32: lane handles cols lane and lane+32).
// ---------------------------------------------------------------------------
__global__ void lgcn_init(const float* __restrict__ ego,
                          float* __restrict__ ego_norm,
                          float* __restrict__ x,
                          float* __restrict__ acc) {
    int row  = blockIdx.x * 8 + (threadIdx.x >> 5);
    int lane = threadIdx.x & 31;
    if (row >= N_TOT) return;
    size_t base = (size_t)row * EMB;
    float a = ego[base + lane], b = ego[base + lane + 32];
    float ss = a * a + b * b;
#pragma unroll
    for (int off = 16; off; off >>= 1) ss += __shfl_xor(ss, off, 32);
    if (lane == 0) ego_norm[row] = sqrtf(ss);
    x[base + lane] = a;       x[base + lane + 32] = b;
    acc[base + lane] = a;     acc[base + lane + 32] = b;
}

// ---------------------------------------------------------------------------
// Stage 4a: zero the SpMM output buffer
// ---------------------------------------------------------------------------
__global__ void lgcn_zero(float* __restrict__ y, int n) {
    int idx = blockIdx.x * blockDim.x + threadIdx.x;
    if (idx < n) y[idx] = 0.f;
}

// ---------------------------------------------------------------------------
// Stage 4b: SpMM scatter:  y[rows[e]] += vals[e] * x[cols[e]]
// 16 threads per edge; each thread handles 4 columns with a float4 gather
// (global_load_b128) + 4 global_atomic_add_f32.  x/y (38 MB each) are
// L2-resident on MI455X (192 MB L2), so this is L2-throughput bound.
// ---------------------------------------------------------------------------
__global__ void lgcn_spmm_scatter(const int*   __restrict__ rows,
                                  const int*   __restrict__ cols,
                                  const float* __restrict__ vals,
                                  const float* __restrict__ x,
                                  float*       __restrict__ y) {
    long long t = (long long)blockIdx.x * blockDim.x + threadIdx.x;
    long long e = t >> 4;
    if (e >= NNZ) return;
    int c4 = (int)(t & 15) * 4;
    int r  = rows[e];
    int cl = cols[e];
    float v = vals[e];
    const float4 xv = *(const float4*)(x + (size_t)cl * EMB + c4);
    float* yp = y + (size_t)r * EMB + c4;
    atomicAdd(yp + 0, v * xv.x);
    atomicAdd(yp + 1, v * xv.y);
    atomicAdd(yp + 2, v * xv.z);
    atomicAdd(yp + 3, v * xv.w);
}

// ---------------------------------------------------------------------------
// Stage 4c: cosine re-weight + accumulate:
//   w = <y,ego> / max(|y| * |ego|, eps);  x = w*y;  acc += x
// One wave per row.
// ---------------------------------------------------------------------------
__global__ void lgcn_cosine_update(const float* __restrict__ y,
                                   const float* __restrict__ ego,
                                   const float* __restrict__ ego_norm,
                                   float* __restrict__ x,
                                   float* __restrict__ acc) {
    int row  = blockIdx.x * 8 + (threadIdx.x >> 5);
    int lane = threadIdx.x & 31;
    if (row >= N_TOT) return;
    size_t base = (size_t)row * EMB;
    float y0 = y[base + lane],  y1 = y[base + lane + 32];
    float e0 = ego[base + lane], e1 = ego[base + lane + 32];
    float dot = y0 * e0 + y1 * e1;
    float nn  = y0 * y0 + y1 * y1;
#pragma unroll
    for (int off = 16; off; off >>= 1) {
        dot += __shfl_xor(dot, off, 32);
        nn  += __shfl_xor(nn,  off, 32);
    }
    float w = dot / fmaxf(sqrtf(nn) * ego_norm[row], 1e-8f);
    float nx0 = w * y0, nx1 = w * y1;
    x[base + lane]        = nx0;
    x[base + lane + 32]   = nx1;
    acc[base + lane]     += nx0;
    acc[base + lane + 32] += nx1;
}

// ---------------------------------------------------------------------------
extern "C" void kernel_launch(void* const* d_in, const int* in_sizes, int n_in,
                              void* d_out, int out_size, void* d_ws, size_t ws_size,
                              hipStream_t stream) {
    const float* user_fea = (const float*)d_in[0];
    const float* item_fea = (const float*)d_in[1];
    const float* prompt   = (const float*)d_in[2];
    const float* mlp_w    = (const float*)d_in[3];
    const float* mlp_b    = (const float*)d_in[4];
    const int*   adj_rows = (const int*)d_in[5];
    const int*   adj_cols = (const int*)d_in[6];
    const float* adj_vals = (const float*)d_in[7];

    float* acc = (float*)d_out;                         // [N_TOT, EMB] = tuple concat

    // workspace partition (all 256B-aligned since N_TOT*EMB*4 is)
    float* ego      = (float*)d_ws;                     // 38.4 MB
    float* x        = ego + (size_t)N_TOT * EMB;        // 38.4 MB
    float* y        = x   + (size_t)N_TOT * EMB;        // 38.4 MB
    float* ego_norm = y   + (size_t)N_TOT * EMB;        // 0.6 MB

    // ego: user part
    lgcn_user_embed<<<(N_USER * EMB + 255) / 256, 256, 0, stream>>>(
        user_fea, prompt, ego);

    // ego: item part (WMMA GEMM + tanh); 3125 row-tiles, 8 waves/block
    lgcn_item_mlp_wmma<<<(N_ITEM / 16 + 7) / 8, 256, 0, stream>>>(
        item_fea, mlp_w, mlp_b, ego + (size_t)N_USER * EMB);

    // norms + seed x and acc
    lgcn_init<<<(N_TOT + 7) / 8, 256, 0, stream>>>(ego, ego_norm, x, acc);

    for (int layer = 0; layer < N_LAYERS; ++layer) {
        lgcn_zero<<<(N_TOT * EMB + 255) / 256, 256, 0, stream>>>(y, N_TOT * EMB);
        lgcn_spmm_scatter<<<(int)(((long long)NNZ * 16 + 255) / 256), 256, 0, stream>>>(
            adj_rows, adj_cols, adj_vals, x, y);
        lgcn_cosine_update<<<(N_TOT + 7) / 8, 256, 0, stream>>>(
            y, ego, ego_norm, x, acc);
    }
}
